// MultiHeadAttention_40097814675891
// MI455X (gfx1250) — matrix operand order repaired and checked by reading, not compiled, not run on hardware
//
#include <hip/hip_runtime.h>

// MHA forward for MI455X (gfx1250, wave32).
//   kernel 0: one-shot fp32 -> bf16 conversion of x and Wq/Wk/Wv
//   kernel 1: QKV GEMM on the bf16 WMMA pipe, 16x64 per wave, double-buffered
//             with sched_barrier-enforced load/WMMA overlap
//   kernel 2: flash attention, Q frags hoisted, K/V frags pipelined across
//             the softmax barriers (loads in flight while VALU runs)

#define HID   1024
#define NB    2
#define SEQ   2048
#define NH    16
#define HD    64
#define ROWS  (NB*SEQ)          // 4096 total query rows

typedef __attribute__((ext_vector_type(16))) __bf16 v16bf;
typedef __attribute__((ext_vector_type(8)))  float  v8f;

union Frag {
  v16bf          v;
  unsigned short s[16];
  uint4          q[2];
};

// round-to-nearest-even fp32 -> bf16 (raw bits)
__device__ __forceinline__ unsigned short f2bf(float f) {
  union { float f; unsigned u; } c; c.f = f;
  unsigned u = c.u;
  u += 0x7FFFu + ((u >> 16) & 1u);
  return (unsigned short)(u >> 16);
}

__device__ __forceinline__ void cvt4(unsigned short* d, float4 f) {
  d[0] = f2bf(f.x); d[1] = f2bf(f.y); d[2] = f2bf(f.z); d[3] = f2bf(f.w);
}

// A-fragment (16x32, MxK): lane -> row lane%16; 16 contiguous k at
// k = 16*(lane/16).
__device__ __forceinline__ void loadA(Frag& a, const unsigned short* row,
                                      int kk, int half) {
  a.q[0] = ((const uint4*)(row + kk + 16 * half))[0];
  a.q[1] = ((const uint4*)(row + kk + 16 * half + 8))[0];
}
// B-fragment (32x16, KxN): lane -> col lane%16; 16 contiguous k at 16*(lane/16).
__device__ __forceinline__ void loadB(Frag& b, const unsigned short* p) {
  b.q[0] = ((const uint4*)p)[0];
  b.q[1] = ((const uint4*)p)[1];
}

__device__ __forceinline__ v8f wmma_bf16(const Frag& a, const Frag& b, v8f c) {
  return __builtin_amdgcn_wmma_f32_16x16x32_bf16(
      false, a.v, false, b.v, (short)0, c, false, false);
}

// ---------------------------------------------------------------------------
// Kernel 0: fp32 -> bf16 pre-conversion (grid-stride, 8 elems/thread/iter).
// ---------------------------------------------------------------------------
__global__ __launch_bounds__(256) void cvt_bf16(const float* __restrict__ src,
                                                unsigned short* __restrict__ dst,
                                                int n8) {
  int i = blockIdx.x * blockDim.x + threadIdx.x;
  const int stride = gridDim.x * blockDim.x;
  for (; i < n8; i += stride) {
    const float4* p = (const float4*)(src + (size_t)i * 8);
    const float4 f0 = p[0], f1 = p[1];
    union { uint4 q; unsigned short s[8]; } o;
    cvt4(o.s + 0, f0); cvt4(o.s + 4, f1);
    ((uint4*)(dst + (size_t)i * 8))[0] = o.q;
  }
}

// ---------------------------------------------------------------------------
// Kernel 1: QKV GEMM (bf16 in, bf16 out).  grid = (ROWS/32, HID/256, 3),
// block = 256 (8 waves, 2x4).  Wave tile 16 rows x 64 cols; K-loop 1024/32.
// Double-buffered fragments; sched_barrier(0) pins the next-step loads above
// the current-step WMMAs so the backend cannot sink them and recycle VGPRs
// (which would force s_wait_loadcnt 0 before every WMMA).
// Outputs: Q,K as [B,H,S,D]; V transposed [B,H,D,S].
// ---------------------------------------------------------------------------
__global__ __launch_bounds__(256) void qkv_gemm(
    const unsigned short* __restrict__ xb,   // [4096,1024] bf16
    const unsigned short* __restrict__ wb,   // [3][1024,1024] bf16
    const float* __restrict__ bq, const float* __restrict__ bk,
    const float* __restrict__ bv,
    unsigned short* __restrict__ Qb, unsigned short* __restrict__ Kb,
    unsigned short* __restrict__ Vt)
{
  const int tid  = threadIdx.x;
  const int lane = tid & 31, wave = tid >> 5;
  const int half = lane >> 4, lm = lane & 15;
  const int wr = wave >> 2, wc = wave & 3;
  const int row0 = blockIdx.x * 32 + wr * 16;
  const int col0 = blockIdx.y * 256 + wc * 64;
  const int mat  = blockIdx.z;

  const unsigned short* w    = wb + (size_t)mat * HID * HID;
  const float*          bias = (mat == 0) ? bq : (mat == 1) ? bk : bv;

  const unsigned short* xrow = xb + (size_t)(row0 + lm) * HID;
  const unsigned short* wrow = w  + (size_t)(col0 + lm) * HID + 16 * half;

  v8f acc[4] = {{}, {}, {}, {}};

  Frag a0, b0[4];
  loadA(a0, xrow, 0, half);
  #pragma unroll
  for (int t = 0; t < 4; ++t) loadB(b0[t], wrow + (size_t)t * 16 * HID);

  #pragma unroll 2
  for (int kk = 0; kk < HID - 32; kk += 32) {
    // issue next k-step's 10 loads into fresh registers ...
    Frag a1, b1[4];
    loadA(a1, xrow, kk + 32, half);
    #pragma unroll
    for (int t = 0; t < 4; ++t)
      loadB(b1[t], wrow + (size_t)t * 16 * HID + kk + 32);

    // ... and fence the scheduler: loads stay above, WMMAs below.
    __builtin_amdgcn_sched_barrier(0);

    #pragma unroll
    for (int t = 0; t < 4; ++t) acc[t] = wmma_bf16(a0, b0[t], acc[t]);

    a0 = a1;
    #pragma unroll
    for (int t = 0; t < 4; ++t) b0[t] = b1[t];
  }
  #pragma unroll
  for (int t = 0; t < 4; ++t) acc[t] = wmma_bf16(a0, b0[t], acc[t]);

  // C/D layout: lane covers col N = lane%16 in every VGPR; row M = r + 8*half.
  #pragma unroll
  for (int t = 0; t < 4; ++t) {
    const int c = col0 + t * 16 + lm, hh = c >> 6, dd = c & (HD - 1);
    const float bval = bias[c];
    #pragma unroll
    for (int r = 0; r < 8; ++r) {
      const int grow = row0 + r + 8 * half;
      const int bidx = grow >> 11;           // /SEQ
      const int srow = grow & (SEQ - 1);
      const unsigned short bf = f2bf(acc[t][r] + bval);
      if (mat == 0)
        Qb[((((size_t)bidx * NH + hh) * SEQ) + srow) * HD + dd] = bf;
      else if (mat == 1)
        Kb[((((size_t)bidx * NH + hh) * SEQ) + srow) * HD + dd] = bf;
      else
        Vt[((((size_t)bidx * NH + hh) * HD + dd) * SEQ) + srow] = bf;
    }
  }
}

// ---------------------------------------------------------------------------
// Kernel 2: flash attention.  grid = (SEQ/32, NB*NH), block = 256 (8 waves).
// WG owns 32 query rows; streams keys in blocks of 64 (32 iterations).
// Q fragments are loop-invariant (hoisted).  K/V fragments for the NEXT block
// are issued right after the score WMMAs and fenced, so the global loads are
// in flight across both softmax barriers.
// ---------------------------------------------------------------------------
__global__ __launch_bounds__(256) void flash_attn(
    const unsigned short* __restrict__ Qb,
    const unsigned short* __restrict__ Kb,
    const unsigned short* __restrict__ Vt,
    float* __restrict__ out)
{
  __shared__ float          sS[32 * 68];   // scores fp32, padded stride
  __shared__ unsigned short sP[32 * 72];   // probs bf16, padded stride
  __shared__ float sM[32], sL[32], sScale[32];

  const int tid  = threadIdx.x;
  const int lane = tid & 31, wave = tid >> 5;
  const int half = lane >> 4, lm = lane & 15;
  const int q0 = blockIdx.x * 32;
  const int bh = blockIdx.y;

  const unsigned short* Qh  = Qb + (size_t)bh * SEQ * HD;
  const unsigned short* Kh  = Kb + (size_t)bh * SEQ * HD;
  const unsigned short* Vth = Vt + (size_t)bh * HD * SEQ;

  const int sq  = wave >> 2, sk = wave & 3;   // score-phase wave grid
  const int wq_ = wave >> 2, wd = wave & 3;   // O-phase wave grid
  const int srowq = tid >> 3, j = tid & 7;    // softmax: 8 threads per row

  if (tid < 32) { sM[tid] = -1e30f; sL[tid] = 0.0f; }
  v8f oacc = {};

  // loop-invariant Q fragments (rows q0 + sq*16 + lane%16, d = 0..63)
  Frag qa[2];
  {
    const unsigned short* qr = Qh + (size_t)(q0 + sq * 16 + lm) * HD;
    loadA(qa[0], qr, 0, half);
    loadA(qa[1], qr, 32, half);
  }
  // base pointers for K (score B-operand) and V^T (O B-operand)
  const unsigned short* krow = Kh + (size_t)(sk * 16 + lm) * HD + 16 * half;
  const unsigned short* vrow = Vth + (size_t)(wd * 16 + lm) * SEQ + 16 * half;

  // preload first key block's K and V fragments
  Frag kf[2], vf[2];
  loadB(kf[0], krow);                   // kk = 0   (keys kb + sk*16 .. +15)
  loadB(kf[1], krow + 32);              // kk = 32
  loadB(vf[0], vrow);                   // keys 0..31
  loadB(vf[1], vrow + 32);              // keys 32..63

  __syncthreads();

  #pragma unroll 2
  for (int kb = 0; kb < SEQ; kb += 64) {
    // ---- scores: S = Q_tile x K_block^T (fragments already resident) ----
    v8f sacc = {};
    sacc = wmma_bf16(qa[0], kf[0], sacc);
    sacc = wmma_bf16(qa[1], kf[1], sacc);

    // ---- issue next block's K/V loads; they fly across the barriers ----
    int kbn = kb + 64;
    if (kbn >= SEQ) kbn = 0;            // harmless reload on last iteration
    Frag kn[2], vn[2];
    loadB(kn[0], krow + (size_t)kbn * HD);
    loadB(kn[1], krow + (size_t)kbn * HD + 32);
    loadB(vn[0], vrow + kbn);
    loadB(vn[1], vrow + kbn + 32);
    __builtin_amdgcn_sched_barrier(0);  // keep them issued before the barrier

    #pragma unroll
    for (int r = 0; r < 8; ++r)
      sS[(sq * 16 + r + 8 * half) * 68 + sk * 16 + lm] = sacc[r];
    __syncthreads();

    // ---- online softmax over this 64-key block ----
    {
      float e[8];
      const float* rowp = sS + srowq * 68 + j * 8;
      float mloc = -1e30f;
      #pragma unroll
      for (int t = 0; t < 8; ++t) { e[t] = rowp[t]; mloc = fmaxf(mloc, e[t]); }
      #pragma unroll
      for (int off = 1; off < 8; off <<= 1) mloc = fmaxf(mloc, __shfl_xor(mloc, off));
      const float mold  = sM[srowq];
      const float mnew  = fmaxf(mold, mloc);
      const float scale = __expf(mold - mnew);
      float lsum = 0.0f;
      #pragma unroll
      for (int t = 0; t < 8; ++t) { e[t] = __expf(e[t] - mnew); lsum += e[t]; }
      #pragma unroll
      for (int off = 1; off < 8; off <<= 1) lsum += __shfl_xor(lsum, off);
      if (j == 0) { sM[srowq] = mnew; sL[srowq] = sL[srowq] * scale + lsum; sScale[srowq] = scale; }
      unsigned short* pp = sP + srowq * 72 + j * 8;
      #pragma unroll
      for (int t = 0; t < 8; ++t) pp[t] = f2bf(e[t]);
    }
    __syncthreads();

    // ---- O = O*scale + P x V_block ----
    #pragma unroll
    for (int r = 0; r < 8; ++r)
      oacc[r] *= sScale[wq_ * 16 + r + 8 * half];
    {
      Frag pa;
      const unsigned short* pr = sP + (wq_ * 16 + lm) * 72;
      loadA(pa, pr, 0, half);
      oacc = wmma_bf16(pa, vf[0], oacc);
      loadA(pa, pr, 32, half);
      oacc = wmma_bf16(pa, vf[1], oacc);
    }

    // rotate double buffers
    kf[0] = kn[0]; kf[1] = kn[1];
    vf[0] = vn[0]; vf[1] = vn[1];
    // next iteration's first __syncthreads orders sP/sScale reuse
  }

  // ---- epilogue: normalize by row sum, write [B,S,H*D] fp32 ----
  const int b_ = bh >> 4, h_ = bh & 15;
  #pragma unroll
  for (int r = 0; r < 8; ++r) {
    const int lrow = wq_ * 16 + r + 8 * half;
    const int srow = q0 + lrow;
    out[((size_t)b_ * SEQ + srow) * HID + h_ * HD + wd * 16 + lm] =
        oacc[r] / sL[lrow];
  }
}

// ---------------------------------------------------------------------------
extern "C" void kernel_launch(void* const* d_in, const int* in_sizes, int n_in,
                              void* d_out, int out_size, void* d_ws, size_t ws_size,
                              hipStream_t stream) {
  const float* x  = (const float*)d_in[0];
  // d_in[1] = masked (mask branch is a no-op in the reference)
  const float* wq = (const float*)d_in[2];
  const float* bq = (const float*)d_in[3];
  const float* wk = (const float*)d_in[4];
  const float* bk = (const float*)d_in[5];
  const float* wv = (const float*)d_in[6];
  const float* bv = (const float*)d_in[7];
  float* out = (float*)d_out;

  // workspace layout (bf16 = unsigned short):
  //   xbf [4096,1024]  8 MB | wbf [3][1024,1024]  6 MB
  //   Qb,Kb [B,H,S,D], Vt [B,H,D,S]  3 x 8 MB
  unsigned short* xbf = (unsigned short*)d_ws;
  unsigned short* wbf = xbf + (size_t)ROWS * HID;
  unsigned short* Qb  = wbf + (size_t)3 * HID * HID;
  unsigned short* Kb  = Qb + (size_t)ROWS * HID;
  unsigned short* Vt  = Kb + (size_t)ROWS * HID;

  cvt_bf16<<<512, 256, 0, stream>>>(x,  xbf,                 ROWS * HID / 8);
  cvt_bf16<<<512, 256, 0, stream>>>(wq, wbf + 0 * HID * HID, HID * HID / 8);
  cvt_bf16<<<512, 256, 0, stream>>>(wk, wbf + 1 * HID * HID, HID * HID / 8);
  cvt_bf16<<<512, 256, 0, stream>>>(wv, wbf + 2 * HID * HID, HID * HID / 8);

  dim3 g1(ROWS / 32, HID / 256, 3), b1(256);
  qkv_gemm<<<g1, b1, 0, stream>>>(xbf, wbf, bq, bk, bv, Qb, Kb, Vt);

  dim3 g2(SEQ / 32, NB * NH), b2(256);
  flash_attn<<<g2, b2, 0, stream>>>(Qb, Kb, Vt, out);
}
